// SNN_Actor_81123342287275
// MI455X (gfx1250) — matrix operand order repaired and checked by reading, not compiled, not run on hardware
//
#include <hip/hip_runtime.h>
#include <hip/hip_bf16.h>

typedef __attribute__((ext_vector_type(16))) _Float16 v16h;
typedef __attribute__((ext_vector_type(8)))  _Float16 v8h;
typedef __attribute__((ext_vector_type(8)))  float    v8f;
typedef __attribute__((ext_vector_type(4)))  int      v4i;

#define BATCH     4096
#define OBS_DIM   64
#define EN_POP    10
#define IN_POP    640       // OBS_DIM * EN_POP
#define H0        1024
#define H1        1024
#define OUT_POP   320
#define ACT_DIM   32
#define DE_POP    10
#define SPIKE_TS  16

#define ENC_VTH       0.999f
#define NEURON_CDECAY 0.5f
#define NEURON_VTH    0.5f
#define THETA_R       0.021f
#define THETA_S       0.132f
#define THETA_U       0.529f
#define THETA_V       (-0.172f)

#if defined(__gfx1250__) && __has_builtin(__builtin_amdgcn_global_load_async_to_lds_b128)
#define USE_ASYNC_LDS 1
#define GLOBAL_AS __attribute__((address_space(1)))
#define LDS_AS    __attribute__((address_space(3)))
#else
#define USE_ASYNC_LDS 0
#endif

// ---------------------------------------------------------------------------
// Elementwise kernels
// ---------------------------------------------------------------------------
__global__ void k_f32_to_f16(const float* __restrict__ in, _Float16* __restrict__ out, int n) {
    int i = blockIdx.x * blockDim.x + threadIdx.x;
    if (i < n) out[i] = (_Float16)in[i];
}

// pop_act[m, j*EN_POP+p] = exp(-0.5*((obs[m,j]-mean)/std)^2)
__global__ void k_pop_act(const float* __restrict__ obs, const float* __restrict__ mean,
                          const float* __restrict__ stdv, float* __restrict__ pop, int total) {
    int i = blockIdx.x * blockDim.x + threadIdx.x;
    if (i >= total) return;
    int m  = i / IN_POP;
    int jp = i % IN_POP;
    int j  = jp / EN_POP;
    float d = (obs[m * OBS_DIM + j] - mean[jp]) / stdv[jp];
    pop[i] = __expf(-0.5f * d * d);
}

// one encoder integrate-and-fire step; spike emitted as f16 {0,1} (WMMA-ready)
__global__ void k_enc_step(const float* __restrict__ pop, float* __restrict__ volt,
                           _Float16* __restrict__ spk, int total) {
    int i = blockIdx.x * blockDim.x + threadIdx.x;
    if (i >= total) return;
    float v = volt[i] + pop[i];
    float s = (v > ENC_VTH) ? 1.0f : 0.0f;
    volt[i] = v - s * ENC_VTH;
    spk[i]  = (_Float16)s;
}

// decoder: per-action grouped dot + tanh
__global__ void k_decode(const float* __restrict__ acc, const float* __restrict__ dec_w,
                         const float* __restrict__ dec_b, float* __restrict__ out, int total) {
    int i = blockIdx.x * blockDim.x + threadIdx.x;
    if (i >= total) return;
    int m = i >> 5;       // / ACT_DIM
    int a = i & 31;       // % ACT_DIM
    const float* p = acc   + (size_t)m * OUT_POP + a * DE_POP;
    const float* w = dec_w + a * DE_POP;
    float s = dec_b[a];
#pragma unroll
    for (int q = 0; q < DE_POP; ++q)
        s += p[q] * (1.0f / (float)SPIKE_TS) * w[q];
    out[i] = tanhf(s);
}

// ---------------------------------------------------------------------------
// Fused spike-GEMM + quadratic-LIF kernel.
//   C[m,n] = sum_k A[m,k]*W[n,k]  (A = f16 spikes, W = f16 weights, f32 acc)
// Block: 256 threads = 8 wave32. Block tile M=256 x N=64.
// Wave w owns M-rows [mBlk+32w, +32): 2 A fragments x 4 B fragments = 8 WMMA
// per 32-K step; every B fragment (LDS) is reused across two A fragments.
// Weights double-buffered through LDS via gfx1250 async global->LDS DMA.
// ---------------------------------------------------------------------------
template <int N, int K>
__global__ __launch_bounds__(256) void k_gemm_lif(
    const _Float16* __restrict__ A,    // [BATCH x K] input spikes
    const _Float16* __restrict__ W,    // [N x K] f16 weights (row n = neuron n)
    const float*    __restrict__ bias, // [N]
    float* __restrict__ cst, float* __restrict__ vst, float* __restrict__ ust,
    _Float16* __restrict__ sst,        // [BATCH x N] spike state (read s_{t-1}, write s_t)
    float* __restrict__ acc_out)       // optional spike-count accumulator, or nullptr
{
    __shared__ _Float16 lds[2][64 * 32];

    const int tid  = threadIdx.x;
    const int lane = tid & 31;
    const int wave = tid >> 5;
    const int nBlk = blockIdx.x * 64;
    const int mBlk = blockIdx.y * 256;
    const int m0   = mBlk + wave * 32;

    const int lrow = lane & 15;          // row-in-tile (A: M, B: N)
    const int aOff = (lane >> 4) * 8;    // A K-chunk base within 32 (ISA A layout)
    const int bOff = (lane >> 4) * 8;    // B K-chunk base within 32 (same as A layout)

    // LDS staging coords: 64 rows x 4 chunks of 8 halves, one b128 per thread
    const int srow = tid >> 2;
    const int schk = (tid & 3) * 8;
    const _Float16* wsrc = W + (size_t)(nBlk + srow) * K + schk;
    _Float16* const ldst[2] = { &lds[0][srow * 32 + schk], &lds[1][srow * 32 + schk] };

    v8f acc[2][4];
#pragma unroll
    for (int h = 0; h < 2; ++h)
#pragma unroll
        for (int j = 0; j < 4; ++j) acc[h][j] = (v8f)(0.0f);

    const int KT = K >> 5;

    // prologue: stage k-tile 0 into buffer 0
#if USE_ASYNC_LDS
    __builtin_amdgcn_global_load_async_to_lds_b128(
        (GLOBAL_AS v4i*)wsrc, (LDS_AS v4i*)ldst[0], 0, 0);
    asm volatile("s_wait_asynccnt 0x0" ::: "memory");
#else
    *(v8h*)ldst[0] = *(const v8h*)wsrc;
#endif
    __syncthreads();

    const _Float16* arow0 = A + (size_t)(m0 + lrow) * K + aOff;
    const _Float16* arow1 = arow0 + (size_t)16 * K;

#pragma unroll 2
    for (int kt = 0; kt < KT; ++kt) {
        const int cur = kt & 1;
        if (kt + 1 < KT) {
#if USE_ASYNC_LDS
            __builtin_amdgcn_global_load_async_to_lds_b128(
                (GLOBAL_AS v4i*)(wsrc + (size_t)(kt + 1) * 32),
                (LDS_AS v4i*)ldst[cur ^ 1], 0, 0);
#else
            const v8h wv = *(const v8h*)(wsrc + (size_t)(kt + 1) * 32);
            *(v8h*)ldst[cur ^ 1] = wv;
#endif
        }
        if (kt + 2 < KT) {  // prefetch spike stream two K-tiles ahead
            __builtin_prefetch(arow0 + (kt + 2) * 32, 0, 1);
            __builtin_prefetch(arow1 + (kt + 2) * 32, 0, 1);
        }

        // A fragments: lane holds K chunks [aOff, aOff+8) and [aOff+16, aOff+24)
        v8h a0lo = *(const v8h*)(arow0 + kt * 32);
        v8h a0hi = *(const v8h*)(arow0 + kt * 32 + 16);
        v8h a1lo = *(const v8h*)(arow1 + kt * 32);
        v8h a1hi = *(const v8h*)(arow1 + kt * 32 + 16);
        v16h af0, af1;
#pragma unroll
        for (int i = 0; i < 8; ++i) {
            af0[i] = a0lo[i]; af0[8 + i] = a0hi[i];
            af1[i] = a1lo[i]; af1[8 + i] = a1hi[i];
        }

        const _Float16* lbase = &lds[cur][0];
#pragma unroll
        for (int j = 0; j < 4; ++j) {
            // B fragment: lane holds column nBlk+16j+lrow, K chunks [bOff, bOff+8) and [bOff+16, bOff+24)
            const _Float16* brow = lbase + (16 * j + lrow) * 32 + bOff;
            v8h blo = *(const v8h*)(brow);
            v8h bhi = *(const v8h*)(brow + 16);
            v16h bf;
#pragma unroll
            for (int i = 0; i < 8; ++i) { bf[i] = blo[i]; bf[8 + i] = bhi[i]; }
            acc[0][j] = __builtin_amdgcn_wmma_f32_16x16x32_f16(
                false, af0, false, bf, (short)0, acc[0][j], false, false);
            acc[1][j] = __builtin_amdgcn_wmma_f32_16x16x32_f16(
                false, af1, false, bf, (short)0, acc[1][j], false, false);
        }
#if USE_ASYNC_LDS
        asm volatile("s_wait_asynccnt 0x0" ::: "memory");
#endif
        __syncthreads();
    }

    // Epilogue: quadratic LIF on the C-fragment layout
    // lanes 0-15: M=r, N=lane ; lanes 16-31: M=8+r, N=lane-16
#pragma unroll
    for (int h = 0; h < 2; ++h) {
#pragma unroll
        for (int j = 0; j < 4; ++j) {
            const int n = nBlk + 16 * j + lrow;
            const float bn = bias[n];
#pragma unroll
            for (int r = 0; r < 8; ++r) {
                const int m = m0 + 16 * h + (lane >> 4) * 8 + r;
                const size_t idx = (size_t)m * N + n;
                float c  = cst[idx] * NEURON_CDECAY + acc[h][j][r] + bn;
                float sp = (float)sst[idx];
                float v  = vst[idx] * (1.0f - sp) + THETA_R * sp;
                float u  = ust[idx] + sp * THETA_S;
                float dv = v * v - v - u + c;
                float du = THETA_V * v + THETA_U * u;
                v += dv; u += du;
                float sn = (v > NEURON_VTH) ? 1.0f : 0.0f;
                cst[idx] = c; vst[idx] = v; ust[idx] = u;
                sst[idx] = (_Float16)sn;
                if (acc_out) acc_out[idx] += sn;
            }
        }
    }
}

// ---------------------------------------------------------------------------
// Host launcher
// ---------------------------------------------------------------------------
extern "C" void kernel_launch(void* const* d_in, const int* in_sizes, int n_in,
                              void* d_out, int out_size, void* d_ws, size_t ws_size,
                              hipStream_t stream) {
    const float* obs  = (const float*)d_in[0];
    const float* mean = (const float*)d_in[1];
    const float* stdv = (const float*)d_in[2];
    const float* W0   = (const float*)d_in[3];
    const float* b0   = (const float*)d_in[4];
    const float* W1   = (const float*)d_in[5];
    const float* b1   = (const float*)d_in[6];
    const float* Wout = (const float*)d_in[7];
    const float* bout = (const float*)d_in[8];
    const float* dw   = (const float*)d_in[9];
    const float* db   = (const float*)d_in[10];
    float* out = (float*)d_out;

    char* ws = (char*)d_ws;
    size_t off = 0;
    auto alloc = [&](size_t bytes) -> void* {
        void* p = ws + off;
        off += (bytes + 255) & ~(size_t)255;
        return p;
    };

    _Float16* wf0 = (_Float16*)alloc((size_t)H0 * IN_POP * 2);
    _Float16* wf1 = (_Float16*)alloc((size_t)H1 * H0 * 2);
    _Float16* wf2 = (_Float16*)alloc((size_t)OUT_POP * H1 * 2);
    float* pop      = (float*)alloc((size_t)BATCH * IN_POP * 4);
    float* enc_volt = (float*)alloc((size_t)BATCH * IN_POP * 4);
    _Float16* spk   = (_Float16*)alloc((size_t)BATCH * IN_POP * 2);
    float* c0 = (float*)alloc((size_t)BATCH * H0 * 4);
    float* v0 = (float*)alloc((size_t)BATCH * H0 * 4);
    float* u0 = (float*)alloc((size_t)BATCH * H0 * 4);
    _Float16* s0 = (_Float16*)alloc((size_t)BATCH * H0 * 2);
    float* c1 = (float*)alloc((size_t)BATCH * H1 * 4);
    float* v1 = (float*)alloc((size_t)BATCH * H1 * 4);
    float* u1 = (float*)alloc((size_t)BATCH * H1 * 4);
    _Float16* s1 = (_Float16*)alloc((size_t)BATCH * H1 * 2);
    float* c2 = (float*)alloc((size_t)BATCH * OUT_POP * 4);
    float* v2 = (float*)alloc((size_t)BATCH * OUT_POP * 4);
    float* u2 = (float*)alloc((size_t)BATCH * OUT_POP * 4);
    _Float16* s2 = (_Float16*)alloc((size_t)BATCH * OUT_POP * 2);
    float* act_sum = (float*)alloc((size_t)BATCH * OUT_POP * 4);
    (void)in_sizes; (void)n_in; (void)out_size; (void)ws_size;

    // zero all persistent state each call (harness does not re-zero)
    (void)hipMemsetAsync(enc_volt, 0, (size_t)BATCH * IN_POP * 4, stream);
    (void)hipMemsetAsync(c0, 0, (size_t)BATCH * H0 * 4, stream);
    (void)hipMemsetAsync(v0, 0, (size_t)BATCH * H0 * 4, stream);
    (void)hipMemsetAsync(u0, 0, (size_t)BATCH * H0 * 4, stream);
    (void)hipMemsetAsync(s0, 0, (size_t)BATCH * H0 * 2, stream);
    (void)hipMemsetAsync(c1, 0, (size_t)BATCH * H1 * 4, stream);
    (void)hipMemsetAsync(v1, 0, (size_t)BATCH * H1 * 4, stream);
    (void)hipMemsetAsync(u1, 0, (size_t)BATCH * H1 * 4, stream);
    (void)hipMemsetAsync(s1, 0, (size_t)BATCH * H1 * 2, stream);
    (void)hipMemsetAsync(c2, 0, (size_t)BATCH * OUT_POP * 4, stream);
    (void)hipMemsetAsync(v2, 0, (size_t)BATCH * OUT_POP * 4, stream);
    (void)hipMemsetAsync(u2, 0, (size_t)BATCH * OUT_POP * 4, stream);
    (void)hipMemsetAsync(s2, 0, (size_t)BATCH * OUT_POP * 2, stream);
    (void)hipMemsetAsync(act_sum, 0, (size_t)BATCH * OUT_POP * 4, stream);

    // weight fp32 -> f16 (once per call; deterministic)
    {
        int n = H0 * IN_POP;
        k_f32_to_f16<<<(n + 255) / 256, 256, 0, stream>>>(W0, wf0, n);
        n = H1 * H0;
        k_f32_to_f16<<<(n + 255) / 256, 256, 0, stream>>>(W1, wf1, n);
        n = OUT_POP * H1;
        k_f32_to_f16<<<(n + 255) / 256, 256, 0, stream>>>(Wout, wf2, n);
    }

    // gaussian population activation (time-invariant)
    {
        int n = BATCH * IN_POP;
        k_pop_act<<<(n + 255) / 256, 256, 0, stream>>>(obs, mean, stdv, pop, n);
    }

    const dim3 gH(H0 / 64, BATCH / 256);       // (16, 16)
    const dim3 gO(OUT_POP / 64, BATCH / 256);  // (5, 16)
    const int encN = BATCH * IN_POP;

    for (int t = 0; t < SPIKE_TS; ++t) {
        k_enc_step<<<(encN + 255) / 256, 256, 0, stream>>>(pop, enc_volt, spk, encN);
        k_gemm_lif<H0, IN_POP><<<gH, 256, 0, stream>>>(
            spk, wf0, b0, c0, v0, u0, s0, nullptr);
        k_gemm_lif<H1, H0><<<gH, 256, 0, stream>>>(
            s0, wf1, b1, c1, v1, u1, s1, nullptr);
        k_gemm_lif<OUT_POP, H1><<<gO, 256, 0, stream>>>(
            s1, wf2, bout, c2, v2, u2, s2, act_sum);
    }

    {
        int n = BATCH * ACT_DIM;
        k_decode<<<(n + 255) / 256, 256, 0, stream>>>(act_sum, dw, db, out, n);
    }
}
